// NCP_71004399337966
// MI455X (gfx1250) — matrix operand order repaired and checked by reading, not compiled, not run on hardware
//
#include <hip/hip_runtime.h>
#include <hip/hip_bf16.h>

typedef __attribute__((ext_vector_type(16))) _Float16 v16h;
typedef __attribute__((ext_vector_type(8)))  float    v8f;

#define B_DIM 256
#define T_DIM 1024
#define BT (B_DIM * T_DIM)
#define M_TILES (BT / 16)

// ---------------------------------------------------------------------------
// Kernel 1: sens[b,t,:] = relu(x @ d1_w + d1_b) @ d2_w + d2_b  via WMMA
// One wave computes a 16x32 output tile. GEMM1 (K=2) is folded into the
// A-fragment build; GEMM2 (K=128) is 4x v_wmma_f32_16x16x32_f16 per n-tile.
// ---------------------------------------------------------------------------
__global__ __launch_bounds__(256) void ncp_dense_wmma(
    const float* __restrict__ x,     // [BT,2]
    const float* __restrict__ d1_w,  // [2,128]
    const float* __restrict__ d1_b,  // [128]
    const float* __restrict__ d2_w,  // [128,32]
    const float* __restrict__ d2_b,  // [32]
    float* __restrict__ sens)        // [BT,32]
{
    __shared__ float s_w0[128], s_w1[128], s_b1[128];
    const int tid = threadIdx.x;
    if (tid < 128) {
        s_w0[tid] = d1_w[tid];
        s_w1[tid] = d1_w[128 + tid];
        s_b1[tid] = d1_b[tid];
    }
    __syncthreads();

    const int lane   = tid & 31;
    const int lm     = lane & 15;          // M-row (A) / N-col (B,C,D) within tile
    const int hi     = lane >> 4;          // lane half select
    const int wave   = (blockIdx.x * blockDim.x + tid) >> 5;
    const int nwaves = (gridDim.x * blockDim.x) >> 5;

    // B fragments (d2_w as f16), invariant across m-tiles: [ktile][ntile]
    // Layout: lanes 0-15 hold K=0..15 (2 packed per VGPR), lanes 16-31 K=16..31.
    v16h bfrag[4][2];
#pragma unroll
    for (int kt = 0; kt < 4; ++kt)
#pragma unroll
        for (int nt = 0; nt < 2; ++nt)
#pragma unroll
            for (int r2 = 0; r2 < 16; ++r2) {
                int k = kt * 32 + hi * 16 + r2;
                int n = nt * 16 + lm;
                bfrag[kt][nt][r2] = (_Float16)d2_w[k * 32 + n];
            }

    const float bias0 = d2_b[lm];
    const float bias1 = d2_b[16 + lm];

    for (int mt = wave; mt < M_TILES; mt += nwaves) {
        const int mrow = mt * 16 + lm;     // this lane's A row
        const float x0 = x[mrow * 2 + 0];
        const float x1 = x[mrow * 2 + 1];

        v8f c0 = {}, c1 = {};
#pragma unroll
        for (int kt = 0; kt < 4; ++kt) {
            // A fragment: relu(x @ d1_w + b) restricted to this lane's K set.
            // lanes 0-15: K=0..15; lanes 16-31: K=16..31.
            v16h a;
#pragma unroll
            for (int r2 = 0; r2 < 16; ++r2) {
                int k = kt * 32 + hi * 16 + r2;
                float h = x0 * s_w0[k] + x1 * s_w1[k] + s_b1[k];
                a[r2] = (_Float16)fmaxf(h, 0.0f);
            }
            c0 = __builtin_amdgcn_wmma_f32_16x16x32_f16(
                     false, a, false, bfrag[kt][0], (short)0, c0, false, false);
            c1 = __builtin_amdgcn_wmma_f32_16x16x32_f16(
                     false, a, false, bfrag[kt][1], (short)0, c1, false, false);
        }
        // D layout: lane = N column; VGPR r = row r (lanes 0-15) / r+8 (16-31).
#pragma unroll
        for (int r = 0; r < 8; ++r) {
            int row = mt * 16 + r + hi * 8;
            sens[row * 32 + lm]      = c0[r] + bias0;
            sens[row * 32 + 16 + lm] = c1[r] + bias1;
        }
    }
}

// ---------------------------------------------------------------------------
// Kernel 2: LTC recurrent scan. One wave per batch element; lane = unit (32).
// Recurrent param columns in registers, sensory params in LDS, state
// broadcast via v_readlane, sigmoid via v_exp_f32 + v_rcp_f32.
// ---------------------------------------------------------------------------
__device__ __forceinline__ float bcast32(float v, int i) {
    return __int_as_float(__builtin_amdgcn_readlane(__float_as_int(v), i));
}
__device__ __forceinline__ float fsig(float xx) {
    // sigmoid(x) = 1 / (1 + 2^(-x*log2 e))
    return __builtin_amdgcn_rcpf(1.0f + __builtin_amdgcn_exp2f(-1.44269504088896340736f * xx));
}

__global__ __launch_bounds__(256) void ncp_ltc_scan(
    const float* __restrict__ sens,       // [B,T,32] from kernel 1
    const float* __restrict__ input_w,  const float* __restrict__ input_b,
    const float* __restrict__ output_w, const float* __restrict__ output_b,
    const float* __restrict__ gleak,    const float* __restrict__ vleak,
    const float* __restrict__ cm,
    const float* __restrict__ w_syn,    const float* __restrict__ mu,
    const float* __restrict__ sigma,    const float* __restrict__ erev,
    const float* __restrict__ sensory_w,  const float* __restrict__ sensory_mu,
    const float* __restrict__ sensory_sigma, const float* __restrict__ sensory_erev,
    const int* __restrict__ spm,        const int* __restrict__ ssm,
    float* __restrict__ out)              // [B]
{
    __shared__ float s_sw[1024], s_smu[1024], s_ssig[1024], s_serev[1024];
    const int tid = threadIdx.x;
    for (int e = tid; e < 1024; e += 256) {
        s_sw[e]    = sensory_w[e] * (float)ssm[e];   // fold sensory mask
        s_smu[e]   = sensory_mu[e];
        s_ssig[e]  = sensory_sigma[e];
        s_serev[e] = sensory_erev[e];
    }
    __syncthreads();

    const int lane = tid & 31;                     // post-synaptic unit j
    const int b    = blockIdx.x * 8 + (tid >> 5);  // batch element

    // Column j of recurrent params, mask folded into weight -> registers.
    float rw[32], rmu[32], rsig[32], rerev[32];
#pragma unroll
    for (int i = 0; i < 32; ++i) {
        int e = i * 32 + lane;
        rw[i]    = w_syn[e] * (float)spm[e];
        rmu[i]   = mu[e];
        rsig[i]  = sigma[e];
        rerev[i] = erev[e];
    }

    const float gl   = gleak[lane];
    const float glvl = gl * vleak[lane];
    const float cmt  = cm[lane] * 6.0f;            // cm * ODE_UNFOLDS
    const float iw   = input_w[lane];
    const float ib   = input_b[lane];

    const float* __restrict__ hrow = sens + (size_t)b * T_DIM * 32;
    float v = 0.0f;

    for (int t = 0; t < T_DIM; ++t) {
        const float inp = hrow[t * 32 + lane] * iw + ib;
        __builtin_prefetch(hrow + (t + 1) * 32 + lane, 0, 1);

        // Sensory contribution (once per timestep)
        float ns = 0.0f, dsn = 0.0f;
#pragma unroll
        for (int i = 0; i < 32; ++i) {
            float xi = bcast32(inp, i);
            int   e  = i * 32 + lane;
            float g  = fsig(s_ssig[e] * (xi - s_smu[e]));
            float a  = s_sw[e] * g;
            ns  = fmaf(a, s_serev[e], ns);
            dsn += a;
        }

        // 6 ODE unfolds
#pragma unroll
        for (int u = 0; u < 6; ++u) {
            float n = ns, d = dsn;
#pragma unroll
            for (int i = 0; i < 32; ++i) {
                float vi = bcast32(v, i);
                float g  = fsig(rsig[i] * (vi - rmu[i]));
                float a  = rw[i] * g;
                n = fmaf(a, rerev[i], n);
                d += a;
            }
            float num = cmt * v + glvl + n;
            float den = cmt + gl + d + 1e-8f;
            v = num * __builtin_amdgcn_rcpf(den);
        }
    }

    if (lane == 0) out[b] = v * output_w[0] + output_b[0];
}

// ---------------------------------------------------------------------------
extern "C" void kernel_launch(void* const* d_in, const int* in_sizes, int n_in,
                              void* d_out, int out_size, void* d_ws, size_t ws_size,
                              hipStream_t stream) {
    const float* x        = (const float*)d_in[0];
    const float* d1_w     = (const float*)d_in[1];
    const float* d1_b     = (const float*)d_in[2];
    const float* d2_w     = (const float*)d_in[3];
    const float* d2_b     = (const float*)d_in[4];
    const float* input_w  = (const float*)d_in[5];
    const float* input_b  = (const float*)d_in[6];
    const float* output_w = (const float*)d_in[7];
    const float* output_b = (const float*)d_in[8];
    const float* gleak    = (const float*)d_in[9];
    const float* vleak    = (const float*)d_in[10];
    const float* cm       = (const float*)d_in[11];
    const float* w_syn    = (const float*)d_in[12];
    const float* mu       = (const float*)d_in[13];
    const float* sigma    = (const float*)d_in[14];
    const float* erev     = (const float*)d_in[15];
    const float* sens_w   = (const float*)d_in[16];
    const float* sens_mu  = (const float*)d_in[17];
    const float* sens_sig = (const float*)d_in[18];
    const float* sens_erev= (const float*)d_in[19];
    const int*   spm      = (const int*)d_in[20];
    const int*   ssm      = (const int*)d_in[21];

    float* sens = (float*)d_ws;            // [BT,32] = 32 MB scratch
    float* out  = (float*)d_out;           // [256]

    // Phase 1: dense stack via WMMA. 2048 waves x 8 m-tiles each.
    ncp_dense_wmma<<<256, 256, 0, stream>>>(x, d1_w, d1_b, d2_w, d2_b, sens);

    // Phase 2: 256 LTC chains, one wave each (8 waves/block x 32 blocks).
    ncp_ltc_scan<<<32, 256, 0, stream>>>(sens,
        input_w, input_b, output_w, output_b,
        gleak, vleak, cm,
        w_syn, mu, sigma, erev,
        sens_w, sens_mu, sens_sig, sens_erev,
        spm, ssm, out);
}
